// GINNetTianshouPPO_ACTOR_44976897524024
// MI455X (gfx1250) — matrix operand order repaired and checked by reading, not compiled, not run on hardware
//
#include <hip/hip_runtime.h>
#include <math.h>

// ---------------- problem constants (from reference) ----------------
#define Bc   8
#define Nc   10000
#define Ec   160000
#define INC  64
#define HIDc 128
#define OUTC 64
#define Mrows (Bc * Nc)   // 80000, divisible by 128

// ---------------- CDNA5 vector types ----------------
typedef __attribute__((ext_vector_type(16))) __bf16       v16bf;
typedef __attribute__((ext_vector_type(8)))  float        v8f;
typedef __attribute__((ext_vector_type(4)))  unsigned int v4u;
typedef __attribute__((ext_vector_type(8)))  unsigned int v8u;

union FragB16 { v4u q[2]; v16bf v; };

// ---------------- agg init: agg = (1+eps) * x (conv0 only) ----------------
__global__ __launch_bounds__(256)
void scale_copy_kernel(const float* __restrict__ x, float* __restrict__ y,
                       const float* __restrict__ eps, long long n4) {
    long long i = (long long)blockIdx.x * blockDim.x + threadIdx.x;
    if (i >= n4) return;
    float s = 1.0f + *eps;
    float4 v = ((const float4*)x)[i];
    v.x *= s; v.y *= s; v.z *= s; v.w *= s;
    ((float4*)y)[i] = v;
}

// ---------------- edge scatter-add: agg[dst] += x[src] ----------------
template<int D>
__global__ __launch_bounds__(256)
void scatter_add_kernel(const float* __restrict__ X, float* __restrict__ AGG,
                        const int* __restrict__ edges) {
    constexpr int TPE = D / 4;                       // threads per edge
    long long idx = (long long)blockIdx.x * blockDim.x + threadIdx.x;
    long long total = (long long)Bc * Ec * TPE;
    if (idx >= total) return;
    int c4       = (int)(idx % TPE);
    long long eg = idx / TPE;
    int b        = (int)(eg / Ec);
    int e        = (int)(eg % Ec);
    int src = edges[(size_t)b * 2 * Ec + e];
    int dst = edges[(size_t)b * 2 * Ec + Ec + e];
    const float4 v = *(const float4*)(X + ((size_t)b * Nc + src) * D + c4 * 4);
    float* o = AGG + ((size_t)b * Nc + dst) * D + c4 * 4;
    atomicAdd(o + 0, v.x);
    atomicAdd(o + 1, v.y);
    atomicAdd(o + 2, v.z);
    atomicAdd(o + 3, v.w);
}

// ---------------- WMMA GEMM + bias + leaky: Y = act(X[M,K] @ W[K,N] + b) ----
// 256 threads = 8 waves; each wave owns 16 rows, all NOUT columns.
// A tile (128 x K fp32) staged global->LDS by the Tensor Data Mover with
// hardware row padding (pad_interval = one row, pad_amount = 4 dwords).
// All A fragments are converted to bf16 once, before the K loop, so the
// inner loop is pure ds_load_b128(B) + v_wmma.
// Optionally also writes Y2 = (1+eps2)*act(...) (fused agg-init of next conv).
template<int K, int NOUT>
__global__ __launch_bounds__(256)
void gemm_bias_act_kernel(const float* __restrict__ X, const float* __restrict__ W,
                          const float* __restrict__ bias, float* __restrict__ Y,
                          float* __restrict__ Y2, const float* __restrict__ eps2,
                          float slope) {
    constexpr int KP  = K + 8;                       // bf16 W rows: +16B pad
    constexpr int KPf = K + 4;                       // f32 A rows: +16B TDM pad
    constexpr int NT  = NOUT / 16;                   // output tiles per wave
    constexpr int KC  = K / 32;                      // k-chunks
    __shared__ __bf16 sWt[NOUT * KP];                // W transposed: [n][k] bf16
    __shared__ float  sAf[128  * KPf];               // A tile (TDM): [m][k] f32
    __shared__ float  sB[NOUT];

    const int tid = threadIdx.x;
    const int rowBase = blockIdx.x * 128;

    // ---- TDM: DMA the 128 x K fp32 A-tile into LDS (one wave issues) ----
    if (tid < 32) {
        unsigned           lds = (unsigned)(size_t)(void*)&sAf[0];
        unsigned long long ga  = (unsigned long long)(size_t)(X + (size_t)rowBase * K);
        constexpr unsigned PADI = (K == 128) ? 6u : 5u;  // pad every 2^(i+1)=K dwords
        v4u g0; v8u g1;
        g0[0] = 1u;                                   // count=1, load, no gather
        g0[1] = lds;                                  // lds_addr
        g0[2] = (unsigned)ga;                         // global_addr[31:0]
        g0[3] = (unsigned)(ga >> 32) | (2u << 30);    // global_addr[56:32] | type=2
        g1[0] = (2u << 16)                            // data_size = 4 bytes
              | (1u << 20)                            // pad_enable
              | (PADI << 22)                          // pad_interval = K dwords
              | (3u << 25);                           // pad_amount  = 4 dwords
        g1[1] = ((unsigned)K) << 16;                  // tensor_dim0[15:0]
        g1[2] = ((unsigned)Mrows & 0xFFFFu) << 16;    // dim0 hi16=0 | tensor_dim1 lo16
        g1[3] = ((unsigned)Mrows >> 16)               // tensor_dim1 hi16
              | (((unsigned)K) << 16);                // tile_dim0 = K
        g1[4] = 128u;                                 // tile_dim1 = 128, tile_dim2 = 0
        g1[5] = (unsigned)K;                          // tensor_dim0_stride = K
        g1[6] = 0u;
        g1[7] = 0u;
        asm volatile("tensor_load_to_lds %0, %1" :: "s"(g0), "s"(g1) : "memory");
        __builtin_amdgcn_s_wait_tensorcnt(0);
    }

    // ---- stage W [K][NOUT] fp32 -> sWt [NOUT][KP] bf16 (transpose) ----
    for (int idx = tid * 4; idx < K * NOUT; idx += 256 * 4) {
        int k = idx / NOUT, n = idx % NOUT;           // n % 4 == 0
        float4 w4 = *(const float4*)(W + idx);
        sWt[(n + 0) * KP + k] = (__bf16)w4.x;
        sWt[(n + 1) * KP + k] = (__bf16)w4.y;
        sWt[(n + 2) * KP + k] = (__bf16)w4.z;
        sWt[(n + 3) * KP + k] = (__bf16)w4.w;
    }
    if (tid < NOUT) sB[tid] = bias[tid];
    __syncthreads();

    const int wv = tid >> 5;          // wave id 0..7
    const int lane = tid & 31;
    const int lh = lane >> 4;         // lane half (0/1)
    const int ln = lane & 15;
    const int r0 = wv * 16;           // first row of this wave's strip

    // ---- build all A fragments up front (native f32->bf16 converts) ----
    // ISA 16-bit A 16x32 layout:
    // lanes 0-15: halves 0..7 = K 0..7, halves 8..15 = K 16..23
    // lanes 16-31: halves 0..7 = K 8..15, halves 8..15 = K 24..31
    FragB16 af[KC];
#pragma unroll
    for (int kc = 0; kc < KC; ++kc) {
        const float* ap = &sAf[(r0 + ln) * KPf + kc * 32 + lh * 8];
        float4 f0 = *(const float4*)(ap);
        float4 f1 = *(const float4*)(ap + 4);
        float4 f2 = *(const float4*)(ap + 16);
        float4 f3 = *(const float4*)(ap + 20);
        v16bf v;
        v[0]  = (__bf16)f0.x; v[1]  = (__bf16)f0.y; v[2]  = (__bf16)f0.z; v[3]  = (__bf16)f0.w;
        v[4]  = (__bf16)f1.x; v[5]  = (__bf16)f1.y; v[6]  = (__bf16)f1.z; v[7]  = (__bf16)f1.w;
        v[8]  = (__bf16)f2.x; v[9]  = (__bf16)f2.y; v[10] = (__bf16)f2.z; v[11] = (__bf16)f2.w;
        v[12] = (__bf16)f3.x; v[13] = (__bf16)f3.y; v[14] = (__bf16)f3.z; v[15] = (__bf16)f3.w;
        af[kc].v = v;
    }

    v8f acc[NT];
#pragma unroll
    for (int t = 0; t < NT; ++t) { v8f z = {}; acc[t] = z; }

#pragma unroll
    for (int kc = 0; kc < KC; ++kc) {
#pragma unroll
        for (int t = 0; t < NT; ++t) {
            // B fragment: lane = column n, halves 0..7 = K 8*lh.., halves 8..15 = K 16+8*lh..
            FragB16 bf;
            const __bf16* bp = &sWt[(t * 16 + ln) * KP + kc * 32 + lh * 8];
            bf.q[0] = *(const v4u*)bp;
            bf.q[1] = *(const v4u*)(bp + 16);
            acc[t] = __builtin_amdgcn_wmma_f32_16x16x32_bf16(
                false, af[kc].v, false, bf.v, (short)0, acc[t], false, false);
        }
    }

    // epilogue: C/D layout -> m = r + 8*lh, n = ln
    const bool dup = (Y2 != nullptr);
    const float s2 = dup ? (1.0f + *eps2) : 0.0f;
#pragma unroll
    for (int t = 0; t < NT; ++t) {
        int col = t * 16 + ln;
        float bv = sB[col];
#pragma unroll
        for (int r = 0; r < 8; ++r) {
            int m = r + lh * 8;
            float v = acc[t][r] + bv;
            v = v > 0.0f ? v : v * slope;
            size_t idx = (size_t)(rowBase + r0 + m) * NOUT + col;
            Y[idx] = v;
            if (dup) Y2[idx] = v * s2;   // fused (1+eps)*x agg-init for next conv
        }
    }
}

// ---------------- lin3: logits[row] = x[row,0:64] . w + b ----------------
__global__ __launch_bounds__(256)
void lin3_kernel(const float* __restrict__ X, const float* __restrict__ w,
                 const float* __restrict__ b, float* __restrict__ out, int M) {
    __shared__ float sw[OUTC];
    if (threadIdx.x < OUTC) sw[threadIdx.x] = w[threadIdx.x];
    __syncthreads();
    int row = blockIdx.x * blockDim.x + threadIdx.x;
    if (row >= M) return;
    const float* xr = X + (size_t)row * OUTC;
    float acc = 0.0f;
#pragma unroll
    for (int k = 0; k < OUTC; k += 4) {
        float4 v = *(const float4*)(xr + k);
        acc += v.x * sw[k] + v.y * sw[k + 1] + v.z * sw[k + 2] + v.w * sw[k + 3];
    }
    out[row] = acc + b[0];
}

// ---------------- masked softmax over N per batch ----------------
__global__ __launch_bounds__(1024)
void softmax_kernel(const float* __restrict__ logits, const int* __restrict__ mask,
                    float* __restrict__ out) {
    const int b = blockIdx.x, tid = threadIdx.x;
    __shared__ float red[1024];
    const float* lg = logits + (size_t)b * Nc;
    const int*   mk = mask   + (size_t)b * Nc;

    float lmax = -INFINITY;
    for (int n = tid; n < Nc; n += 1024) {
        float v = (mk[n] != 0) ? lg[n] : -INFINITY;
        lmax = fmaxf(lmax, v);
    }
    red[tid] = lmax; __syncthreads();
    for (int s = 512; s > 0; s >>= 1) {
        if (tid < s) red[tid] = fmaxf(red[tid], red[tid + s]);
        __syncthreads();
    }
    float m = red[0]; __syncthreads();

    float lsum = 0.0f;
    for (int n = tid; n < Nc; n += 1024)
        lsum += (mk[n] != 0) ? __expf(lg[n] - m) : 0.0f;
    red[tid] = lsum; __syncthreads();
    for (int s = 512; s > 0; s >>= 1) {
        if (tid < s) red[tid] += red[tid + s];
        __syncthreads();
    }
    float inv = 1.0f / red[0];

    for (int n = tid; n < Nc; n += 1024) {
        float v = (mk[n] != 0) ? __expf(lg[n] - m) : 0.0f;
        out[(size_t)b * Nc + n] = v * inv;
    }
}

// ---------------- host-side orchestration ----------------
struct Lin { const float* w; const float* b; };

extern "C" void kernel_launch(void* const* d_in, const int* in_sizes, int n_in,
                              void* d_out, int out_size, void* d_ws, size_t ws_size,
                              hipStream_t stream) {
    // Robust input binding: handle both insertion-order and sorted-key (jax tree)
    // flattening of the nested inputs dict.
    const bool sorted_mode = (in_sizes[0] == Bc * 2 * Ec);   // edges first => sorted

    const float* gnodes; const int* edges; const int* mask;
    if (sorted_mode) {
        edges  = (const int*)d_in[0];
        gnodes = (const float*)d_in[1];
        mask   = (const int*)d_in[2];
    } else {
        gnodes = (const float*)d_in[0];
        edges  = (const int*)d_in[1];
        mask   = (const int*)d_in[2];
    }
    int p = 3;
    auto takeLin = [&](void) -> Lin {
        Lin l;
        if (sorted_mode) { l.b = (const float*)d_in[p++]; l.w = (const float*)d_in[p++]; }
        else             { l.w = (const float*)d_in[p++]; l.b = (const float*)d_in[p++]; }
        return l;
    };
    const float* eps[4]; Lin cl1[4], cl2[4];
    for (int i = 0; i < 4; ++i) {
        eps[i] = (const float*)d_in[p++];   // "eps" sorts before "l1","l2" either way
        cl1[i] = takeLin();
        cl2[i] = takeLin();
    }
    Lin lin1 = takeLin(), lin2 = takeLin(), lin3 = takeLin();

    // workspace: 3 ping-pong activation buffers [80000 x 128] + logits
    char* ws = (char*)d_ws;
    const size_t bufSz = (size_t)Mrows * HIDc * sizeof(float);   // ~41 MB
    float* bufA   = (float*)(ws);                  // agg
    float* bufB   = (float*)(ws + bufSz);          // mlp hidden
    float* bufC   = (float*)(ws + 2 * bufSz);      // layer output / current x
    float* logits = (float*)(ws + 3 * bufSz);

    const dim3 blk(256);
    const int  gemmGrid = Mrows / 128;             // 625

    // ---- conv0 (D = 64 input) ----
    {
        long long n4 = (long long)Mrows * INC / 4;
        scale_copy_kernel<<<(unsigned)((n4 + 255) / 256), blk, 0, stream>>>(gnodes, bufA, eps[0], n4);
        long long tot = (long long)Bc * Ec * (INC / 4);
        scatter_add_kernel<INC><<<(unsigned)((tot + 255) / 256), blk, 0, stream>>>(gnodes, bufA, edges);
        gemm_bias_act_kernel<INC, HIDc><<<gemmGrid, blk, 0, stream>>>(
            bufA, cl1[0].w, cl1[0].b, bufB, nullptr, nullptr, 0.01f);
        // conv output GEMM: leaky∘leaky folds to slope 1e-4; also emit
        // (1+eps1)*x into bufA as the next conv's agg-init (fused scale_copy).
        gemm_bias_act_kernel<HIDc, HIDc><<<gemmGrid, blk, 0, stream>>>(
            bufB, cl2[0].w, cl2[0].b, bufC, bufA, eps[1], 1e-4f);
    }
    // ---- convs 1..3 (D = 128) ----
    for (int i = 1; i < 4; ++i) {
        long long tot = (long long)Bc * Ec * (HIDc / 4);
        scatter_add_kernel<HIDc><<<(unsigned)((tot + 255) / 256), blk, 0, stream>>>(bufC, bufA, edges);
        gemm_bias_act_kernel<HIDc, HIDc><<<gemmGrid, blk, 0, stream>>>(
            bufA, cl1[i].w, cl1[i].b, bufB, nullptr, nullptr, 0.01f);
        float* y2       = (i < 3) ? bufA    : nullptr;
        const float* e2 = (i < 3) ? eps[i + 1] : nullptr;
        gemm_bias_act_kernel<HIDc, HIDc><<<gemmGrid, blk, 0, stream>>>(
            bufB, cl2[i].w, cl2[i].b, bufC, y2, e2, 1e-4f);
    }
    // ---- heads ----
    gemm_bias_act_kernel<HIDc, HIDc><<<gemmGrid, blk, 0, stream>>>(
        bufC, lin1.w, lin1.b, bufB, nullptr, nullptr, 0.01f);
    gemm_bias_act_kernel<HIDc, OUTC><<<gemmGrid, blk, 0, stream>>>(
        bufB, lin2.w, lin2.b, bufA, nullptr, nullptr, 0.01f);
    lin3_kernel<<<(Mrows + 255) / 256, blk, 0, stream>>>(bufA, lin3.w, lin3.b, logits, Mrows);
    softmax_kernel<<<Bc, 1024, 0, stream>>>(logits, mask, (float*)d_out);
}
